// FlatLadder7x7NNUEv2_47519518163391
// MI455X (gfx1250) — matrix operand order repaired and checked by reading, not compiled, not run on hardware
//
#include <hip/hip_runtime.h>
#include <hip/hip_bf16.h>
#include <math.h>

// ---------------------------------------------------------------------------
// FlatLadder7x7NNUE on gfx1250 (CDNA5): WMMA f16 GEMM pipeline.
//   B=8192 boards, 12 chunk-mappings/board, DM=128, DMAP=48.
// ---------------------------------------------------------------------------

typedef __attribute__((ext_vector_type(16))) _Float16 v16h;
typedef __attribute__((ext_vector_type(8)))  _Float16 v8h;
typedef __attribute__((ext_vector_type(8)))  float    v8f;

#define NBOARD   8192
#define SAMP     16            // samples (boards) per mapping block
#define ASTRIDE  136           // 128 channels + 16B pad (halves)
#define QMAXF    (127.0f/128.0f)
#define NWAVES   4             // 128 threads / block in mapping kernel

// ---- packed-weight workspace layout (bytes, per mapping) -------------------
constexpr int OFF_L1W = 0;        // 768 f32  (3,128,2)
constexpr int OFF_L1B = 3072;     // 128 f32
constexpr int OFF_C1W = 3584;     // 16384 f16 (B-frag packed, KSTEPS=4)
constexpr int OFF_C1B = 36352;    // 128 f32  (BN-folded)
constexpr int OFF_L2W = 36864;    // 49152 f16 (KSTEPS=12)
constexpr int OFF_L2B = 135168;   // 128 f32
constexpr int OFF_C2W = 135680;
constexpr int OFF_C2B = 168448;
constexpr int OFF_L3W = 168960;
constexpr int OFF_L3B = 267264;
constexpr int OFF_C3W = 267776;
constexpr int OFF_C3B = 300544;
constexpr int OFF_COW = 301056;   // 6144 f16 (48x128, KSTEPS=4)
constexpr int OFF_COB = 313344;   // 48 f32 (BN-folded)
constexpr int MAPW_BYTES = 313600;
constexpr size_t PARTIAL_OFF = (size_t)2 * MAPW_BYTES;   // f32 [12][8192][48]

constexpr int SMEM_BYTES = 2048 + 2 * 144 * ASTRIDE * 2; // 80384 B dynamic LDS

// ---- input-pointer bundle (flattened setup_inputs() dict order) ------------
//  0: data (8192*2*7*7)
//  1..30:  m1 {l1w,l1b,l2w,l2b,l3w,l3b, [cW,cB,bnG,bnB,bnM,bnV]x3, cow,cob,bnoG,bnoB,bnoM,bnoV}
//  31..60: m2 (same layout)
//  61..65: pol {w1,b1,w2,b2,q}
//  66..73: val {w0,b0,w1,b1,w2,b2,w3,b3}
struct AllPtrs { const float* p[80]; };

// ---- chunk transform tables: chunk = rot90( (t?transpose:id)(window), k ) --
__constant__ int CY0[12] = {0,0,0,0,3,3,3,3, 0,0,3,3};
__constant__ int CX0[12] = {0,0,3,3,3,3,0,0, 0,3,3,0};
__constant__ int CK [12] = {0,0,1,3,2,2,3,1, 1,2,3,0};
__constant__ int CT [12] = {0,1,0,1,0,1,0,1, 0,0,0,0};

// ladder gather: in-pixel index per (tap, out-pixel)
__constant__ int TABL2[12] = {0,1,3,4,  3,4,6,7,  4,5,7,8}; // 3x3 -> 2x2
__constant__ int TABL3[3]  = {0,2,3};                       // 2x2 -> 1x1

// Branch-free exact mish:  tanh(softplus(x)) = (u^2+2u)/(u^2+2u+2), u = e^x.
// One v_exp_f32 + one v_rcp_f32 (explicit, so no IEEE-div expansion);
// exp arg clamped so n stays finite (ratio saturates to 1).
__device__ __forceinline__ float mish(float x) {
    float u = __expf(fminf(x, 40.0f));
    float n = u * (u + 2.0f);
    return x * n * __builtin_amdgcn_rcpf(n + 2.0f);
}

// position of element (k,n) inside B-fragment-packed block stream
//  B operand in this file's contiguous k order, same as A (32Kx16N f16): lane = n%16 + 16*(k/16),
//  half j = k%16; blocks streamed as [ntile][kstep][512 halves].
__device__ __forceinline__ int packB(int n, int k, int ksteps) {
    int nt = n >> 4, nl = n & 15;
    int ks = k >> 5, kr = k & 31;
    return ((nt * ksteps + ks) << 9) + ((nl + (kr & 16)) << 4) + (kr & 15);
}

// ---------------------------------------------------------------------------
// Kernel 1: fold BN, quantize to f16, swizzle weights into B-fragment order.
// ---------------------------------------------------------------------------
__global__ __launch_bounds__(256) void prep_kernel(AllPtrs P, char* __restrict__ ws) {
    const int gid = blockIdx.x * blockDim.x + threadIdx.x;
    const int gsz = gridDim.x * blockDim.x;

    for (int m = 0; m < 2; ++m) {
        const int base = 1 + m * 30;
        char* mw = ws + (size_t)m * MAPW_BYTES;

        // l1 (kept fp32, evaluated in VALU: K=6 is too small for WMMA)
        {
            float* ow = (float*)(mw + OFF_L1W);
            float* ob = (float*)(mw + OFF_L1B);
            const float* w = P.p[base + 0];
            const float* b = P.p[base + 1];
            for (int i = gid; i < 768; i += gsz) ow[i] = w[i];
            for (int i = gid; i < 128; i += gsz) ob[i] = b[i];
        }
        // ladder l2 / l3: (3,128,128) -> K = tap*128 + ic, KSTEPS=12
        const int LW[2] = {OFF_L2W, OFF_L3W};
        const int LB[2] = {OFF_L2B, OFF_L3B};
        for (int t = 0; t < 2; ++t) {
            const float* w = P.p[base + 2 + 2 * t];
            const float* b = P.p[base + 3 + 2 * t];
            _Float16* ow = (_Float16*)(mw + LW[t]);
            float*    ob = (float*)(mw + LB[t]);
            for (int i = gid; i < 49152; i += gsz) {
                int tap = i / 16384, r = i % 16384;
                int oc = r >> 7, ic = r & 127;
                ow[packB(oc, tap * 128 + ic, 12)] = (_Float16)w[i];
            }
            for (int i = gid; i < 128; i += gsz) ob[i] = b[i];
        }
        // 1x1 conv layers c1..c3 with BN folded
        const int CW[3] = {OFF_C1W, OFF_C2W, OFF_C3W};
        const int CB[3] = {OFF_C1B, OFF_C2B, OFF_C3B};
        for (int l = 0; l < 3; ++l) {
            const int wi = base + 6 + 6 * l;
            const float* w  = P.p[wi + 0];
            const float* cb = P.p[wi + 1];
            const float* g  = P.p[wi + 2];
            const float* bt = P.p[wi + 3];
            const float* mu = P.p[wi + 4];
            const float* vr = P.p[wi + 5];
            _Float16* ow = (_Float16*)(mw + CW[l]);
            float*    ob = (float*)(mw + CB[l]);
            for (int i = gid; i < 16384; i += gsz) {
                int oc = i >> 7;
                float s = g[oc] * rsqrtf(vr[oc] + 1e-5f);
                ow[packB(oc, i & 127, 4)] = (_Float16)(w[i] * s);
            }
            for (int i = gid; i < 128; i += gsz) {
                float s = g[i] * rsqrtf(vr[i] + 1e-5f);
                ob[i] = (cb[i] - mu[i]) * s + bt[i];
            }
        }
        // output conv (48x128) with BN folded
        {
            const int wi = base + 24;
            const float* w  = P.p[wi + 0];
            const float* cb = P.p[wi + 1];
            const float* g  = P.p[wi + 2];
            const float* bt = P.p[wi + 3];
            const float* mu = P.p[wi + 4];
            const float* vr = P.p[wi + 5];
            _Float16* ow = (_Float16*)(mw + OFF_COW);
            float*    ob = (float*)(mw + OFF_COB);
            for (int i = gid; i < 6144; i += gsz) {
                int oc = i >> 7;
                float s = g[oc] * rsqrtf(vr[oc] + 1e-5f);
                ow[packB(oc, i & 127, 4)] = (_Float16)(w[i] * s);
            }
            for (int i = gid; i < 48; i += gsz) {
                float s = g[i] * rsqrtf(vr[i] + 1e-5f);
                ob[i] = (cb[i] - mu[i]) * s + bt[i];
            }
        }
    }
}

// ---------------------------------------------------------------------------
// WMMA GEMM stage.  A: activations from LDS (f16, [row][ch], ASTRIDE-padded).
// B: pre-swizzled weights streamed from global (L2-resident).  C: f32 accum.
// POUT==0: direct rows (1x1 conv).  POUT>0: ladder gather (K = 3 taps x 128).
// ---------------------------------------------------------------------------
template <int MT, int NTI, int KS, int POUT, int PIN, bool FINAL>
__device__ __forceinline__ void gemm_stage(const char* __restrict__ mw, int woff, int boff,
                                           const _Float16* __restrict__ aIn,
                                           _Float16* __restrict__ aOut,
                                           float* __restrict__ gOut,
                                           const int* __restrict__ ptab,
                                           int lane, int wave) {
    const _Float16* W    = (const _Float16*)(mw + woff);
    const float*    bias = (const float*)(mw + boff);
    const int m = lane & 15, h = lane >> 4;

    for (int job = wave; job < MT * NTI; job += NWAVES) {   // wave-uniform: EXEC all-1s at WMMA
        const int mt = job / NTI, nt = job - mt * NTI;
        const int row = mt * 16 + m;
        v8f c = {};
        const _Float16* wb = W + (size_t)(nt * KS) * 512 + lane * 16;
#pragma unroll
        for (int ks = 0; ks < KS; ++ks) {
            int inRow, kOff;
            if (POUT > 0) {
                const int tap = ks >> 2;                      // 128 K per tap
                const int s = row / POUT, op = row % POUT;
                inRow = s * PIN + ptab[tap * POUT + op];
                kOff  = (ks & 3) * 32;
            } else { inRow = row; kOff = ks * 32; }
            // A fragment (16x32 f16): halves 0-15 <-> K=16h..16h+15
            const _Float16* ap = aIn + inRow * ASTRIDE + kOff + 16 * h;
            v8h alo = *(const v8h*)ap;          // ds_load_b128
            v8h ahi = *(const v8h*)(ap + 8);    // ds_load_b128
            v16h a;
#pragma unroll
            for (int e = 0; e < 8; ++e) { a[e] = alo[e]; a[e + 8] = ahi[e]; }
            v16h bf = *(const v16h*)(wb + ks * 512);          // 32B contiguous per lane
            c = __builtin_amdgcn_wmma_f32_16x16x32_f16(false, a, false, bf,
                                                       (short)0, c, false, false);
        }
        const int n = nt * 16 + m;
        const float bn = bias[n];
#pragma unroll
        for (int e = 0; e < 8; ++e) {            // C frag: (M = e + 8h, N = lane%16)
            const int r = mt * 16 + e + 8 * h;
            float y = c[e] + bn;
            if (!FINAL) {
                aOut[(size_t)r * ASTRIDE + n] = (_Float16)mish(y);
            } else {
                y = fminf(fmaxf(y, -1.0f), QMAXF);
                y = rintf(y * 128.0f) * 0.0078125f;           // fake_quant, RNE
                gOut[r * 48 + n] = y;
            }
        }
    }
}

// ---------------------------------------------------------------------------
// Kernel 2: one block = 16 boards of one chunk.  Full CNN in LDS + WMMA.
// ---------------------------------------------------------------------------
__global__ __launch_bounds__(128) void mapping_kernel(const float* __restrict__ data,
                                                      char* __restrict__ ws) {
    extern __shared__ char smem[];
    float*    xin  = (float*)smem;                     // [16 s][2 ch][16 pix]
    _Float16* actA = (_Float16*)(smem + 2048);         // up to 144 rows x ASTRIDE
    _Float16* actB = actA + 144 * ASTRIDE;

    const int blk   = blockIdx.x;
    const int chunk = blk >> 9;           // 512 blocks per chunk (8192/16)
    const int b0    = (blk & 511) << 4;
    const int tid   = threadIdx.x;
    const int lane  = tid & 31, wave = tid >> 5;
    const int mapi  = (chunk < 8) ? 0 : 1;
    const char* mw  = ws + (size_t)mapi * MAPW_BYTES;

    // ---- stage 0: gather transformed 4x4 window into LDS --------------------
    const int y0 = CY0[chunk], x0 = CX0[chunk], rk = CK[chunk], tt = CT[chunk];
    for (int i = tid; i < SAMP * 2 * 16; i += blockDim.x) {
        const int s = i >> 5, ch = (i >> 4) & 1, pix = i & 15;
        const int pi = pix >> 2, pj = pix & 3;
        int a, b;                                   // invert rot90^k
        switch (rk) {
            case 0:  a = pi;     b = pj;     break;
            case 1:  a = pj;     b = 3 - pi; break;
            case 2:  a = 3 - pi; b = 3 - pj; break;
            default: a = 3 - pj; b = pi;     break;
        }
        const int yy = tt ? b : a, xx = tt ? a : b;  // invert transpose
        xin[i] = data[(size_t)(b0 + s) * 98 + ch * 49 + (y0 + yy) * 7 + (x0 + xx)];
    }
    __syncthreads();

    // ---- stage 1: ladder l1 (2ch -> 128ch, 4x4 -> 3x3) in VALU + mish -------
    {
        const float* w = (const float*)(mw + OFF_L1W);  // [3][128][2]
        const float* bb = (const float*)(mw + OFF_L1B);
        const int och = tid;                             // 128 threads
        const float w00 = w[och * 2], w01 = w[och * 2 + 1];
        const float w10 = w[256 + och * 2], w11 = w[256 + och * 2 + 1];
        const float w20 = w[512 + och * 2], w21 = w[512 + och * 2 + 1];
        const float bs = bb[och];
        for (int s = 0; s < SAMP; ++s) {
            const float* xs = xin + s * 32;
#pragma unroll
            for (int p = 0; p < 9; ++p) {
                const int oi = p / 3, oj = p % 3;
                const float y = bs
                    + w00 * xs[oi * 4 + oj]           + w01 * xs[16 + oi * 4 + oj]
                    + w10 * xs[(oi + 1) * 4 + oj]     + w11 * xs[16 + (oi + 1) * 4 + oj]
                    + w20 * xs[(oi + 1) * 4 + oj + 1] + w21 * xs[16 + (oi + 1) * 4 + oj + 1];
                actA[(size_t)(s * 9 + p) * ASTRIDE + och] = (_Float16)mish(y);
            }
        }
    }
    __syncthreads();

    float* gOut = (float*)(ws + PARTIAL_OFF) + ((size_t)chunk * NBOARD + b0) * 48;

    // ---- stages 2..7: WMMA GEMM chain ---------------------------------------
    gemm_stage<9, 8, 4, 0, 0, false>(mw, OFF_C1W, OFF_C1B, actA, actB, nullptr, nullptr, lane, wave);
    __syncthreads();
    gemm_stage<4, 8, 12, 4, 9, false>(mw, OFF_L2W, OFF_L2B, actB, actA, nullptr, TABL2, lane, wave);
    __syncthreads();
    gemm_stage<4, 8, 4, 0, 0, false>(mw, OFF_C2W, OFF_C2B, actA, actB, nullptr, nullptr, lane, wave);
    __syncthreads();
    gemm_stage<1, 8, 12, 1, 4, false>(mw, OFF_L3W, OFF_L3B, actB, actA, nullptr, TABL3, lane, wave);
    __syncthreads();
    gemm_stage<1, 8, 4, 0, 0, false>(mw, OFF_C3W, OFF_C3B, actA, actB, nullptr, nullptr, lane, wave);
    __syncthreads();
    gemm_stage<1, 3, 4, 0, 0, true>(mw, OFF_COW, OFF_COB, actB, nullptr, gOut, nullptr, lane, wave);
}

// ---------------------------------------------------------------------------
// Kernel 3: reduce 12 chunk partials per board, policy + value heads.
// ---------------------------------------------------------------------------
__global__ __launch_bounds__(128) void heads_kernel(AllPtrs P,
                                                    const float* __restrict__ partial,
                                                    float* __restrict__ out) {
    __shared__ float sw1[256], sw2[256], sq[784], sb1[16], sb2[16];
    __shared__ float vw[3][1024], vb[3][32], vw3[96], vb3[3];
    const int tid = threadIdx.x;

    for (int i = tid; i < 256; i += 128) { sw1[i] = P.p[61][i]; sw2[i] = P.p[63][i]; }
    for (int i = tid; i < 16;  i += 128) { sb1[i] = P.p[62][i]; sb2[i] = P.p[64][i]; }
    for (int i = tid; i < 784; i += 128) sq[i] = P.p[65][i];
    for (int l = 0; l < 3; ++l) {
        const float* w = P.p[66 + 2 * l];
        const float* b = P.p[67 + 2 * l];
        for (int i = tid; i < 1024; i += 128) vw[l][i] = rintf(w[i] * 128.f) * 0.0078125f;
        for (int i = tid; i < 32;   i += 128) vb[l][i] = rintf(b[i] * 16384.f) * (1.f / 16384.f);
    }
    for (int i = tid; i < 96; i += 128) vw3[i] = rintf(P.p[72][i] * 128.f) * 0.0078125f;
    if (tid < 3) vb3[tid] = rintf(P.p[73][tid] * 16384.f) * (1.f / 16384.f);
    __syncthreads();

    const int b = blockIdx.x * 128 + tid;
    float feat[48];
#pragma unroll
    for (int c = 0; c < 48; ++c) feat[c] = 0.f;
    for (int ch = 0; ch < 12; ++ch) {
        const float* pp = partial + ((size_t)ch * NBOARD + b) * 48;
#pragma unroll
        for (int c = 0; c < 48; ++c) feat[c] += pp[c];
    }
    // policy head
    float h1[16], h2[16];
#pragma unroll
    for (int i = 0; i < 16; ++i) {
        float s = sb1[i];
#pragma unroll
        for (int j = 0; j < 16; ++j) s += feat[j] * sw1[i * 16 + j];
        h1[i] = fmaxf(s, 0.f);
    }
#pragma unroll
    for (int i = 0; i < 16; ++i) {
        float s = sb2[i];
#pragma unroll
        for (int j = 0; j < 16; ++j) s += h1[j] * sw2[i * 16 + j];
        h2[i] = s;
    }
    float* pout = out + (size_t)3 * NBOARD;
    for (int p = 0; p < 49; ++p) {
        float s = 0.f;
#pragma unroll
        for (int i = 0; i < 16; ++i) s += h2[i] * sq[p * 16 + i];
        pout[(size_t)b * 49 + p] = s;
    }
    // value head (clip -> quantized linear, x4)
    float v[32], t[32];
#pragma unroll
    for (int i = 0; i < 32; ++i) v[i] = feat[16 + i];
    for (int l = 0; l < 3; ++l) {
#pragma unroll
        for (int j = 0; j < 32; ++j) t[j] = fminf(fmaxf(v[j], 0.f), QMAXF);
        for (int o = 0; o < 32; ++o) {
            float s = vb[l][o];
#pragma unroll
            for (int j = 0; j < 32; ++j) s += t[j] * vw[l][o * 32 + j];
            v[o] = s;
        }
    }
#pragma unroll
    for (int j = 0; j < 32; ++j) t[j] = fminf(fmaxf(v[j], 0.f), QMAXF);
    for (int o = 0; o < 3; ++o) {
        float s = vb3[o];
#pragma unroll
        for (int j = 0; j < 32; ++j) s += t[j] * vw3[o * 32 + j];
        out[(size_t)b * 3 + o] = s;
    }
}

// ---------------------------------------------------------------------------
extern "C" void kernel_launch(void* const* d_in, const int* in_sizes, int n_in,
                              void* d_out, int out_size, void* d_ws, size_t ws_size,
                              hipStream_t stream) {
    (void)in_sizes; (void)out_size; (void)ws_size;
    AllPtrs ap;
    for (int i = 0; i < 80; ++i) ap.p[i] = (i < n_in) ? (const float*)d_in[i] : nullptr;
    char* ws = (char*)d_ws;
    const float* data = (const float*)d_in[0];
    float* out = (float*)d_out;

    prep_kernel<<<256, 256, 0, stream>>>(ap, ws);
    mapping_kernel<<<12 * (NBOARD / SAMP), 128, SMEM_BYTES, stream>>>(data, ws);
    heads_kernel<<<NBOARD / 128, 128, 0, stream>>>(ap, (const float*)(ws + PARTIAL_OFF), out);
}
